// MoELayer_70342974374191
// MI455X (gfx1250) — matrix operand order repaired and checked by reading, not compiled, not run on hardware
//
#include <hip/hip_runtime.h>
#include <hip/hip_bf16.h>
#include <math.h>

// ---------------- problem constants (fixed by reference) ----------------
constexpr int N  = 4 * 2048;   // B*T = 8192 tokens
constexpr int C  = 1024;
constexpr int E  = 8;
constexpr int H  = 2728;       // int(2*C/3)*4
constexpr int Hp = 2752;       // H padded to multiple of 64 (43 tiles of 64)

typedef __bf16 bf16x8  __attribute__((ext_vector_type(8)));
typedef __bf16 bf16x16 __attribute__((ext_vector_type(16)));
typedef float  f32x8   __attribute__((ext_vector_type(8)));

union FragA { bf16x16 v; bf16x8 h[2]; };

// ---------------- gate: logits -> top-2 softmax -> comb[N,E] ----------------
__global__ __launch_bounds__(256) void moe_gate(const float* __restrict__ x,
                                                const float* __restrict__ gw,
                                                float* __restrict__ comb) {
    __shared__ float sgw[E * C];   // 32 KB
    for (int i = threadIdx.x; i < E * C; i += 256) sgw[i] = gw[i];
    __syncthreads();

    const int wave = threadIdx.x >> 5;
    const int lane = threadIdx.x & 31;
    const int token = blockIdx.x * 8 + wave;
    const float* xr = x + (size_t)token * C;

    float acc[E];
#pragma unroll
    for (int e = 0; e < E; ++e) acc[e] = 0.0f;

    for (int c = lane; c < C; c += 32) {
        float xv = xr[c];
#pragma unroll
        for (int e = 0; e < E; ++e) acc[e] += xv * sgw[e * C + c];
    }
#pragma unroll
    for (int off = 16; off > 0; off >>= 1) {
#pragma unroll
        for (int e = 0; e < E; ++e) acc[e] += __shfl_xor(acc[e], off, 32);
    }

    if (lane == 0) {
        int bi = 0; float bv = acc[0];
#pragma unroll
        for (int e = 1; e < E; ++e) if (acc[e] > bv) { bv = acc[e]; bi = e; }
        int si = -1; float sv = -3.4e38f;
#pragma unroll
        for (int e = 0; e < E; ++e) if (e != bi && acc[e] > sv) { sv = acc[e]; si = e; }
        float t  = expf(sv - bv);          // <= 1
        float wb = 1.0f / (1.0f + t);
        float ws = t * wb;
        float* row = comb + (size_t)token * E;
#pragma unroll
        for (int e = 0; e < E; ++e) row[e] = 0.0f;
        row[bi] = wb;
        row[si] = ws;
    }
}

// ---------------- fp32 -> bf16 converters ----------------
__global__ __launch_bounds__(256) void conv_bf16(const float* __restrict__ in,
                                                 __bf16* __restrict__ out, int n) {
    int i = blockIdx.x * 256 + threadIdx.x;
    if (i < n) out[i] = (__bf16)in[i];
}

// in [H][C] row-major -> out [Hp][C], zero pad rows >= H
__global__ __launch_bounds__(256) void conv_pad_rows(const float* __restrict__ in,
                                                     __bf16* __restrict__ out) {
    int i = blockIdx.x * 256 + threadIdx.x;
    if (i >= Hp * C) return;
    out[i] = (i < H * C) ? (__bf16)in[i] : (__bf16)0.0f;
}

// in [C][H] row-major -> out [C][Hp], zero pad cols >= H
__global__ __launch_bounds__(256) void conv_pad_cols(const float* __restrict__ in,
                                                     __bf16* __restrict__ out) {
    int i = blockIdx.x * 256 + threadIdx.x;
    if (i >= C * Hp) return;
    int r = i / Hp, c = i - r * Hp;
    out[i] = (c < H) ? (__bf16)in[(size_t)r * H + c] : (__bf16)0.0f;
}

// ---------------- GEMM1: h = silu(x@w1^T) * (x@w3^T), bf16 out ----------------
// block = 256 threads = 8 waves arranged 4(M) x 2(N); wave tile 32x32
// block tile: M=128, N=64.  grid = (Hp/64, N/128)
// A tile (x) staged via global_load_async_to_lds_b128, double buffered.
constexpr int LDA = 40;   // padded LDS row stride (elements) -> 80B, 16B aligned
__global__ __launch_bounds__(256) void moe_gemm1(const __bf16* __restrict__ xb,   // [N][C]
                                                 const __bf16* __restrict__ w1b,  // [Hp][C]
                                                 const __bf16* __restrict__ w3b,  // [Hp][C]
                                                 __bf16* __restrict__ hbuf) {     // [N][Hp]
    __shared__ __bf16 sA[2][128 * LDA];   // 2 x 10 KB

    const int tid  = threadIdx.x;
    const int wave = tid >> 5;
    const int lane = tid & 31;
    const int wm = wave >> 1;          // 0..3 (M)
    const int wn = wave & 1;           // 0..1 (N)
    const int hi  = lane >> 4;
    const int l15 = lane & 15;

    const int mblk = blockIdx.y * 128;
    const int nblk = blockIdx.x * 64;

    // --- async A-stage assignment: 2 threads per row, 32B each ---
    const int crow  = tid >> 1;        // 0..127
    const int chalf = tid & 1;         // 0..1
    const __bf16* gsrcA = xb + (size_t)(mblk + crow) * C + chalf * 16;

    auto stage = [&](int buf, int k0) {
        const __bf16* g = gsrcA + k0;
        unsigned int l = (unsigned int)(uintptr_t)(&sA[buf][crow * LDA + chalf * 16]);
        asm volatile("global_load_async_to_lds_b128 %0, %1, off"
                     :: "v"(l), "v"(g) : "memory");
        asm volatile("global_load_async_to_lds_b128 %0, %1, off offset:16"
                     :: "v"(l), "v"(g) : "memory");
    };

    // --- B row pointers (direct global), per 16-wide n fragment ---
    const __bf16* b1row[2];
    const __bf16* b3row[2];
#pragma unroll
    for (int nf = 0; nf < 2; ++nf) {
        int nr = nblk + wn * 32 + nf * 16 + l15;
        b1row[nf] = w1b + (size_t)nr * C + hi * 16;
        b3row[nf] = w3b + (size_t)nr * C + hi * 16;
    }

    f32x8 acc1[2][2] = {};
    f32x8 acc3[2][2] = {};

    constexpr int KSTEPS = C / 32;     // 32
    stage(0, 0);

    for (int kk = 0; kk < KSTEPS; ++kk) {
        const int cur = kk & 1;
        if (kk + 1 < KSTEPS) {
            stage(cur ^ 1, (kk + 1) * 32);
            asm volatile("s_wait_asynccnt 0x2" ::: "memory");
        } else {
            asm volatile("s_wait_asynccnt 0x0" ::: "memory");
        }
        __syncthreads();

        // A fragments from LDS
        FragA fa[2];
#pragma unroll
        for (int mf = 0; mf < 2; ++mf) {
            const __bf16* sa = &sA[cur][(wm * 32 + mf * 16 + l15) * LDA];
            fa[mf].h[0] = *(const bf16x8*)(sa + hi * 16);
            fa[mf].h[1] = *(const bf16x8*)(sa + hi * 16 + 8);
        }
        // B fragments from global
        const int k0 = kk * 32;
        bf16x16 b1[2], b3[2];
#pragma unroll
        for (int nf = 0; nf < 2; ++nf) {
            b1[nf] = *(const bf16x16*)(b1row[nf] + k0);
            b3[nf] = *(const bf16x16*)(b3row[nf] + k0);
        }
#pragma unroll
        for (int mf = 0; mf < 2; ++mf) {
#pragma unroll
            for (int nf = 0; nf < 2; ++nf) {
                acc1[mf][nf] = __builtin_amdgcn_wmma_f32_16x16x32_bf16(
                    false, fa[mf].v, false, b1[nf], (short)0, acc1[mf][nf], false, false);
                acc3[mf][nf] = __builtin_amdgcn_wmma_f32_16x16x32_bf16(
                    false, fa[mf].v, false, b3[nf], (short)0, acc3[mf][nf], false, false);
            }
        }
        __syncthreads();   // protect buffer (cur^1) reuse next iteration
    }

    // epilogue: h = silu(d1) * d3  (D layout: vgpr r -> m = hi*8 + r, n = l15)
#pragma unroll
    for (int mf = 0; mf < 2; ++mf) {
        const int mbase = mblk + wm * 32 + mf * 16 + hi * 8;
#pragma unroll
        for (int nf = 0; nf < 2; ++nf) {
            const int n = nblk + wn * 32 + nf * 16 + l15;
#pragma unroll
            for (int r = 0; r < 8; ++r) {
                float v1 = acc1[mf][nf][r];
                float v3 = acc3[mf][nf][r];
                float s  = v1 / (1.0f + expf(-v1));   // silu
                hbuf[(size_t)(mbase + r) * Hp + n] = (__bf16)(s * v3);
            }
        }
    }
}

// ---------------- GEMM2: out += comb[:,e] * (h @ w2^T) ----------------
// block = 8 waves arranged 4(M) x 2(N); wave tile 32x32; block tile 128x64
// grid = (C/64, N/128)
__global__ __launch_bounds__(256) void moe_gemm2(const __bf16* __restrict__ hbuf, // [N][Hp]
                                                 const __bf16* __restrict__ w2b,  // [C][Hp]
                                                 const float* __restrict__ comb,  // [N][E]
                                                 float* __restrict__ out,         // [N][C]
                                                 int e) {
    const int wave = threadIdx.x >> 5;
    const int lane = threadIdx.x & 31;
    const int wm = wave >> 1;          // 0..3
    const int wn = wave & 1;           // 0..1
    const int hi  = lane >> 4;
    const int l15 = lane & 15;

    const int mblk = blockIdx.y * 128;
    const int nblk = blockIdx.x * 64;

    const __bf16* arow[2];
    const __bf16* brow[2];
#pragma unroll
    for (int mf = 0; mf < 2; ++mf)
        arow[mf] = hbuf + (size_t)(mblk + wm * 32 + mf * 16 + l15) * Hp;
#pragma unroll
    for (int nf = 0; nf < 2; ++nf)
        brow[nf] = w2b + (size_t)(nblk + wn * 32 + nf * 16 + l15) * Hp + hi * 16;

    f32x8 acc[2][2] = {};

    for (int k0 = 0; k0 < Hp; k0 += 32) {
        FragA fa[2];
#pragma unroll
        for (int mf = 0; mf < 2; ++mf) {
            fa[mf].h[0] = *(const bf16x8*)(arow[mf] + k0 + hi * 16);
            fa[mf].h[1] = *(const bf16x8*)(arow[mf] + k0 + hi * 16 + 8);
        }
        bf16x16 fb[2];
#pragma unroll
        for (int nf = 0; nf < 2; ++nf)
            fb[nf] = *(const bf16x16*)(brow[nf] + k0);
#pragma unroll
        for (int mf = 0; mf < 2; ++mf)
#pragma unroll
            for (int nf = 0; nf < 2; ++nf)
                acc[mf][nf] = __builtin_amdgcn_wmma_f32_16x16x32_bf16(
                    false, fa[mf].v, false, fb[nf], (short)0, acc[mf][nf], false, false);
    }

#pragma unroll
    for (int mf = 0; mf < 2; ++mf) {
        const int mbase = mblk + wm * 32 + mf * 16 + hi * 8;
        float wgt[8];
#pragma unroll
        for (int r = 0; r < 8; ++r)
            wgt[r] = comb[(size_t)(mbase + r) * E + e];
#pragma unroll
        for (int nf = 0; nf < 2; ++nf) {
            const int n = nblk + wn * 32 + nf * 16 + l15;
#pragma unroll
            for (int r = 0; r < 8; ++r)
                out[(size_t)(mbase + r) * C + n] += wgt[r] * acc[mf][nf][r];
        }
    }
}

// ---------------- host launcher ----------------
extern "C" void kernel_launch(void* const* d_in, const int* in_sizes, int n_in,
                              void* d_out, int out_size, void* d_ws, size_t ws_size,
                              hipStream_t stream) {
    const float* x  = (const float*)d_in[0];   // [N][C]
    const float* gw = (const float*)d_in[1];   // [E][C]
    const float* w1 = (const float*)d_in[2];   // [E][H][C]
    const float* w2 = (const float*)d_in[3];   // [E][C][H]
    const float* w3 = (const float*)d_in[4];   // [E][H][C]
    float* out = (float*)d_out;                // [N][C]

    auto align256 = [](size_t v) { return (v + 255) & ~size_t(255); };
    char* ws = (char*)d_ws;
    size_t off = 0;
    float*  comb = (float*)(ws + off);  off += align256((size_t)N * E * sizeof(float));
    __bf16* xb   = (__bf16*)(ws + off); off += align256((size_t)N * C * sizeof(__bf16));
    __bf16* w1b  = (__bf16*)(ws + off); off += align256((size_t)Hp * C * sizeof(__bf16));
    __bf16* w3b  = (__bf16*)(ws + off); off += align256((size_t)Hp * C * sizeof(__bf16));
    __bf16* w2b  = (__bf16*)(ws + off); off += align256((size_t)C * Hp * sizeof(__bf16));
    __bf16* hbuf = (__bf16*)(ws + off); off += align256((size_t)N * Hp * sizeof(__bf16));

    hipMemsetAsync(out, 0, (size_t)N * C * sizeof(float), stream);

    moe_gate<<<N / 8, 256, 0, stream>>>(x, gw, comb);
    conv_bf16<<<(N * C + 255) / 256, 256, 0, stream>>>(x, xb, N * C);

    const int nPadRC = (Hp * C + 255) / 256;
    for (int e = 0; e < E; ++e) {
        conv_pad_rows<<<nPadRC, 256, 0, stream>>>(w1 + (size_t)e * H * C, w1b);
        conv_pad_rows<<<nPadRC, 256, 0, stream>>>(w3 + (size_t)e * H * C, w3b);
        conv_pad_cols<<<nPadRC, 256, 0, stream>>>(w2 + (size_t)e * C * H, w2b);

        dim3 g1(Hp / 64, N / 128);
        moe_gemm1<<<g1, 256, 0, stream>>>(xb, w1b, w3b, hbuf);

        dim3 g2(C / 64, N / 128);
        moe_gemm2<<<g2, 256, 0, stream>>>(hbuf, w2b, comb, out, e);
    }
}
